// SparseMLP_83717502534160
// MI455X (gfx1250) — matrix operand order repaired and checked by reading, not compiled, not run on hardware
//
#include <hip/hip_runtime.h>
#include <hip/hip_bf16.h>

#define ALPHA 5.0f
#define BETA  0.5f
#define B_DIM 2048
#define D_DIM 1024
#define H_DIM 32768
#define O_DIM 512
#define S_DIM 32
#define CHUNK 512
#define NCHUNK (H_DIM / CHUNK)   // 64

typedef float  v8f  __attribute__((ext_vector_type(8)));
typedef __bf16 v8bf __attribute__((ext_vector_type(8)));
typedef __bf16 v16bf __attribute__((ext_vector_type(16)));

// ---------------------------------------------------------------- helpers
__device__ __forceinline__ unsigned short f2bf(float f) {
    unsigned u = __float_as_uint(f);
    // round-to-nearest-even truncation to bf16
    unsigned r = (u + 0x7fffu + ((u >> 16) & 1u)) >> 16;
    return (unsigned short)r;
}

__device__ __forceinline__ float uniform01(unsigned long long k) {
    // splitmix64 -> [0,1)
    k += 0x9E3779B97f4A7C15ull;
    k = (k ^ (k >> 30)) * 0xBF58476D1CE4E5B9ull;
    k = (k ^ (k >> 27)) * 0x94D049BB133111EBull;
    k ^= k >> 31;
    return (float)(k >> 40) * (1.0f / 16777216.0f);
}

// Assemble a 16-wide bf16 A-fragment from the two 8-element K-chunks
__device__ __forceinline__ v16bf load_afrag(const unsigned short* arow, int k0, int hs) {
    v8bf lo = *(const v8bf*)(arow + k0 + hs * 16);
    v8bf hi = *(const v8bf*)(arow + k0 + hs * 16 + 8);
    return __builtin_shufflevector(lo, hi,
             0, 1, 2, 3, 4, 5, 6, 7, 8, 9, 10, 11, 12, 13, 14, 15);
}

// ---------------------------------------------------------------- fp32 -> bf16
__global__ void cvt_bf16_kernel(const float* __restrict__ in,
                                unsigned short* __restrict__ out, size_t n) {
    size_t i = (size_t)blockIdx.x * blockDim.x + threadIdx.x;
    size_t stride = (size_t)gridDim.x * blockDim.x;
    for (; i < n; i += stride) out[i] = f2bf(in[i]);
}

// ---------------------------------------------------------------- WMMA GEMM + fused sigmoid
// block = 256 threads (8 waves, 2(M) x 4(N)); wave tile = 32(M) x 64(N):
// 2 A-fragments x 4 B-fragments -> 8 WMMAs per K-step with full fragment reuse.
// Block tile 64 x 256; grid = (B/64, H/256), x fastest over M for L2 W reuse.
// z[m,h] = sum_d x[m,d]*W_in[h,d] + b_in[h];  P[m,h] = sigmoid(ALPHA*(z-BETA))
__global__ __launch_bounds__(256) void gemm_sigmoid_kernel(
        const unsigned short* __restrict__ Xbf,   // B x D bf16 row-major
        const unsigned short* __restrict__ Wbf,   // H x D bf16 row-major
        const float* __restrict__ b_in,           // H
        float* __restrict__ P)                    // B x H
{
    const int lane = threadIdx.x & 31;
    const int wave = threadIdx.x >> 5;
    const int l16  = lane & 15;
    const int hs   = lane >> 4;          // half-wave select (0/1)
    const int wm   = wave >> 2;          // 0..1
    const int wn   = wave & 3;           // 0..3

    const int mBase = blockIdx.x * 64 + wm * 32;
    const int nBase = blockIdx.y * 256 + wn * 64;

    const unsigned short* arow0 = Xbf + (size_t)(mBase + l16) * D_DIM;
    const unsigned short* arow1 = Xbf + (size_t)(mBase + 16 + l16) * D_DIM;
    const unsigned short* brow0 = Wbf + (size_t)(nBase + 0 * 16 + l16) * D_DIM;
    const unsigned short* brow1 = Wbf + (size_t)(nBase + 1 * 16 + l16) * D_DIM;
    const unsigned short* brow2 = Wbf + (size_t)(nBase + 2 * 16 + l16) * D_DIM;
    const unsigned short* brow3 = Wbf + (size_t)(nBase + 3 * 16 + l16) * D_DIM;

    v8f acc[2][4];
    #pragma unroll
    for (int i = 0; i < 2; ++i)
        #pragma unroll
        for (int j = 0; j < 4; ++j)
            acc[i][j] = (v8f){};

    #pragma unroll 2
    for (int k0 = 0; k0 < D_DIM; k0 += 32) {
        v16bf a0 = load_afrag(arow0, k0, hs);
        v16bf a1 = load_afrag(arow1, k0, hs);
        // B fragment: lane holds N fixed, K {hs*16 .. hs*16+15} contiguous
        v16bf b0 = *(const v16bf*)(brow0 + k0 + hs * 16);
        v16bf b1 = *(const v16bf*)(brow1 + k0 + hs * 16);
        v16bf b2 = *(const v16bf*)(brow2 + k0 + hs * 16);
        v16bf b3 = *(const v16bf*)(brow3 + k0 + hs * 16);
        __builtin_prefetch(brow0 + k0 + 128, 0, 1);
        __builtin_prefetch(arow0 + k0 + 128, 0, 1);

        acc[0][0] = __builtin_amdgcn_wmma_f32_16x16x32_bf16(false, a0, false, b0, (short)0, acc[0][0], false, false);
        acc[0][1] = __builtin_amdgcn_wmma_f32_16x16x32_bf16(false, a0, false, b1, (short)0, acc[0][1], false, false);
        acc[0][2] = __builtin_amdgcn_wmma_f32_16x16x32_bf16(false, a0, false, b2, (short)0, acc[0][2], false, false);
        acc[0][3] = __builtin_amdgcn_wmma_f32_16x16x32_bf16(false, a0, false, b3, (short)0, acc[0][3], false, false);
        acc[1][0] = __builtin_amdgcn_wmma_f32_16x16x32_bf16(false, a1, false, b0, (short)0, acc[1][0], false, false);
        acc[1][1] = __builtin_amdgcn_wmma_f32_16x16x32_bf16(false, a1, false, b1, (short)0, acc[1][1], false, false);
        acc[1][2] = __builtin_amdgcn_wmma_f32_16x16x32_bf16(false, a1, false, b2, (short)0, acc[1][2], false, false);
        acc[1][3] = __builtin_amdgcn_wmma_f32_16x16x32_bf16(false, a1, false, b3, (short)0, acc[1][3], false, false);
    }

    // Epilogue: bias + sigmoid + store P
    #pragma unroll
    for (int j = 0; j < 4; ++j) {
        const int   col  = nBase + j * 16 + l16;
        const float bias = b_in[col];
        #pragma unroll
        for (int i = 0; i < 2; ++i) {
            #pragma unroll
            for (int r = 0; r < 8; ++r) {
                int   row = mBase + i * 16 + r + 8 * hs;
                float z   = acc[i][j][r] + bias;
                float pv  = 1.0f / (1.0f + __expf(-ALPHA * (z - BETA)));
                P[(size_t)row * H_DIM + col] = pv;
            }
        }
    }
}

// ---------------------------------------------------------------- per-(row,chunk) partial sums
// grid = (B, NCHUNK), block = 256
__global__ __launch_bounds__(256) void chunk_sum_kernel(
        const float* __restrict__ P, float* __restrict__ chunkSums)
{
    const int row = blockIdx.x, chunk = blockIdx.y;
    const float* base = P + (size_t)row * H_DIM + (size_t)chunk * CHUNK;
    float s = base[threadIdx.x] + base[threadIdx.x + 256];
    __shared__ float red[256];
    red[threadIdx.x] = s;
    __syncthreads();
    for (int off = 128; off > 0; off >>= 1) {
        if ((int)threadIdx.x < off) red[threadIdx.x] += red[threadIdx.x + off];
        __syncthreads();
    }
    if (threadIdx.x == 0) chunkSums[row * NCHUNK + chunk] = red[0];
}

// ---------------------------------------------------------------- multinomial sampling
// grid = B, block = 32 (one wave per row, one sample per lane)
__global__ __launch_bounds__(32) void sample_kernel(
        const float* __restrict__ P,
        const float* __restrict__ chunkSums,
        int*   __restrict__ idxOut,     // B x S
        float* __restrict__ rowSum)     // B
{
    const int row  = blockIdx.x;
    const int lane = threadIdx.x;
    const float* cs = chunkSums + row * NCHUNK;

    float total = 0.0f;
    for (int i = 0; i < NCHUNK; ++i) total += cs[i];

    const float u = uniform01(((unsigned long long)42u << 40)
                              ^ ((unsigned long long)row << 8)
                              ^ (unsigned long long)lane) * total;

    // level 1: find containing chunk (cdf, side='right')
    float cum = 0.0f, before = 0.0f;
    int c = -1;
    for (int i = 0; i < NCHUNK; ++i) {
        float nc = cum + cs[i];
        if (c < 0 && u < nc) { c = i; before = cum; }
        cum = nc;
    }
    if (c < 0) { c = NCHUNK - 1; before = cum - cs[NCHUNK - 1]; }

    // level 2: scan inside chunk
    const float* pr = P + (size_t)row * H_DIM + (size_t)c * CHUNK;
    float run = before;
    int found = -1;
    for (int i = 0; i < CHUNK; ++i) {
        run += pr[i];
        if (run > u) { found = i; break; }
    }
    if (found < 0) found = CHUNK - 1;

    idxOut[row * S_DIM + lane] = c * CHUNK + found;
    if (lane == 0) rowSum[row] = total;
}

// ---------------------------------------------------------------- gather + scaled sum
// grid = B, block = O (512)
__global__ __launch_bounds__(512) void gather_kernel(
        const float* __restrict__ out_weight,   // H x O
        const int*   __restrict__ idxBuf,       // B x S
        const float* __restrict__ rowSum,       // B
        float* __restrict__ out)                // B x O
{
    const int row = blockIdx.x, t = threadIdx.x;
    const int* idx = idxBuf + row * S_DIM;
    float acc = 0.0f;
    #pragma unroll 4
    for (int s = 0; s < S_DIM; ++s)
        acc += out_weight[(size_t)idx[s] * O_DIM + t];
    out[(size_t)row * O_DIM + t] = acc * (rowSum[row] * (1.0f / (float)S_DIM));
}

// ---------------------------------------------------------------- launch
extern "C" void kernel_launch(void* const* d_in, const int* in_sizes, int n_in,
                              void* d_out, int out_size, void* d_ws, size_t ws_size,
                              hipStream_t stream) {
    const float* x          = (const float*)d_in[0];   // B x D
    const float* W_in       = (const float*)d_in[1];   // H x D
    const float* b_in       = (const float*)d_in[2];   // H
    const float* out_weight = (const float*)d_in[3];   // H x O
    float*       out        = (float*)d_out;           // B x O
    (void)in_sizes; (void)n_in; (void)out_size; (void)ws_size;

    // workspace layout
    char* ws = (char*)d_ws;
    unsigned short* Xbf = (unsigned short*)ws;                                   // 4 MB
    unsigned short* Wbf = (unsigned short*)(ws + (size_t)B_DIM * D_DIM * 2);     // 64 MB
    float* P        = (float*)((char*)Wbf + (size_t)H_DIM * D_DIM * 2);          // 256 MB
    float* chunkS   = (float*)((char*)P + (size_t)B_DIM * H_DIM * 4);            // 512 KB
    float* rowSum   = (float*)((char*)chunkS + (size_t)B_DIM * NCHUNK * 4);      // 8 KB
    int*   idxBuf   = (int*)((char*)rowSum + (size_t)B_DIM * 4);                 // 256 KB

    cvt_bf16_kernel<<<2048, 256, 0, stream>>>(x, Xbf, (size_t)B_DIM * D_DIM);
    cvt_bf16_kernel<<<8192, 256, 0, stream>>>(W_in, Wbf, (size_t)H_DIM * D_DIM);

    // grid.x fastest over M so each N-slice of W stays resident in L2
    gemm_sigmoid_kernel<<<dim3(B_DIM / 64, H_DIM / 256), 256, 0, stream>>>(
        Xbf, Wbf, b_in, P);

    chunk_sum_kernel<<<dim3(B_DIM, NCHUNK), 256, 0, stream>>>(P, chunkS);
    sample_kernel<<<B_DIM, 32, 0, stream>>>(P, chunkS, idxBuf, rowSum);
    gather_kernel<<<B_DIM, O_DIM, 0, stream>>>(out_weight, idxBuf, rowSum, out);
}
